// DeformableTransformerCrossAttention_56951266345224
// MI455X (gfx1250) — compile-verified, matrix operand order checked by reading, not run on hardware
//
#include <hip/hip_runtime.h>
#include <hip/hip_bf16.h>
#include <math.h>

typedef __attribute__((ext_vector_type(16))) _Float16 v16h;
typedef __attribute__((ext_vector_type(8)))  _Float16 v8h;
typedef __attribute__((ext_vector_type(8)))  float    v8f;
typedef __attribute__((ext_vector_type(4)))  float    v4f;

// Problem constants (match reference)
constexpr int Bc  = 4;
constexpr int LQ  = 8192;
constexpr int Sc  = 16384;
constexpr int Dc  = 256;
constexpr int NHc = 8;
constexpr int NPc = 4;
constexpr int DHc = 32;   // Dc / NHc
constexpr int Hc  = 128;
constexpr int Wc  = 128;

// ---------------------------------------------------------------------------
// Weight prep: Wt[n][k] = (f16) W[k][n]   (col-major f16 panel, <=128KB,
// L2-resident, reused by every GEMM wave). Makes B fragments contiguous.
// ---------------------------------------------------------------------------
__global__ void transpose_w_f16_kernel(const float* __restrict__ W,  // [256, N]
                                       _Float16* __restrict__ Wt,    // [N, 256]
                                       int N) {
  const int idx = (int)(blockIdx.x * blockDim.x + threadIdx.x);
  if (idx >= N * 256) return;
  const int n = idx >> 8;
  const int k = idx & 255;
  Wt[idx] = (_Float16)W[(size_t)k * N + n];
}

// ---------------------------------------------------------------------------
// WMMA f16 GEMM: C[M,N] = A[M,256] * B[256,N] + bias[N]  (fp32 accumulate)
// B supplied pre-transposed/converted: Bt[n][k] f16.
// One wave computes a 16 x (16*NT) strip; K tiled by 32.
// Fragment layout (CDNA5 ISA 7.12.2, 16-bit A 16x32):
//   lane L: row/col = L%16, half = L/16
//   elems [0..7]  -> K = k0 + 8*half + j        (contiguous run)
//   elems [8..15] -> K = k0 + 16 + 8*half + j-8 (contiguous run)
// Inner loop is phased: issue ALL loads (4 A + 2*NT B, one clause) ->
// pack A while loads land -> back-to-back WMMA chain (no operand overlap,
// distinct accumulators => no hazard NOPs, single progressive wait).
// ---------------------------------------------------------------------------
template <int NT>
__global__ void gemm_f16_wmma_kernel(const float* __restrict__ A,
                                     const _Float16* __restrict__ Bt,
                                     const float* __restrict__ bias,
                                     float* __restrict__ C,
                                     int M, int N) {
  constexpr int K = 256;
  const int wave = (int)((blockIdx.x * blockDim.x + threadIdx.x) >> 5);
  const int lane = (int)(threadIdx.x & 31);
  const int nGroups = N / (16 * NT);
  const int tm = wave / nGroups;
  const int tg = wave % nGroups;
  if (tm * 16 >= M) return;                 // wave-uniform: EXEC stays all-1s

  const int half = lane >> 4;
  const int r    = lane & 15;
  const float* __restrict__ Arow = A + (size_t)(tm * 16 + r) * K + 8 * half;
  const _Float16* __restrict__ Bbase =
      Bt + (size_t)(tg * NT * 16 + r) * K + 8 * half;

  v8f zero = {0.f, 0.f, 0.f, 0.f, 0.f, 0.f, 0.f, 0.f};
  v8f acc[NT];
#pragma unroll
  for (int t = 0; t < NT; ++t) acc[t] = zero;

  for (int k0 = 0; k0 < K; k0 += 32) {
    // ---- phase 1: issue every load for this k-step ----
    const v4f a0 = *(const v4f*)(Arow + k0);
    const v4f a1 = *(const v4f*)(Arow + k0 + 4);
    const v4f a2 = *(const v4f*)(Arow + k0 + 16);
    const v4f a3 = *(const v4f*)(Arow + k0 + 20);
    v8h b0[NT], b1[NT];
#pragma unroll
    for (int t = 0; t < NT; ++t) {
      const _Float16* __restrict__ Bcol = Bbase + (size_t)t * 16 * K + k0;
      b0[t] = *(const v8h*)(Bcol);
      b1[t] = *(const v8h*)(Bcol + 16);
    }
    // ---- phase 2: pack A while loads are in flight ----
    v16h a;
#pragma unroll
    for (int j = 0; j < 4; ++j) {
      a[j]      = (_Float16)a0[j];
      a[4 + j]  = (_Float16)a1[j];
      a[8 + j]  = (_Float16)a2[j];
      a[12 + j] = (_Float16)a3[j];
    }
    // ---- phase 3: back-to-back WMMA chain ----
#pragma unroll
    for (int t = 0; t < NT; ++t) {
      v16h b;
#pragma unroll
      for (int j = 0; j < 8; ++j) {
        b[j]     = b0[t][j];
        b[8 + j] = b1[t][j];
      }
      acc[t] = __builtin_amdgcn_wmma_f32_16x16x32_f16(
          false, a, false, b, (short)0, acc[t], false, false);
    }
  }

#pragma unroll
  for (int t = 0; t < NT; ++t) {
    const int col = (tg * NT + t) * 16 + r;
    const float bv = bias[col];
#pragma unroll
    for (int v = 0; v < 8; ++v) {
      const int m = tm * 16 + v + 8 * half;
      C[(size_t)m * N + col] = acc[t][v] + bv;
    }
  }
}

// ---------------------------------------------------------------------------
// softmax over groups of 4 (NL*NP = 4), in place. One thread per group.
// ---------------------------------------------------------------------------
__global__ void softmax4_kernel(float* __restrict__ a, int n) {
  const int i = (int)(blockIdx.x * blockDim.x + threadIdx.x);
  if (i >= n) return;
  float4 v = ((float4*)a)[i];
  const float m = fmaxf(fmaxf(v.x, v.y), fmaxf(v.z, v.w));
  const float ex = __expf(v.x - m);
  const float ey = __expf(v.y - m);
  const float ez = __expf(v.z - m);
  const float ew = __expf(v.w - m);
  const float inv = 1.0f / (ex + ey + ez + ew);
  ((float4*)a)[i] = make_float4(ex * inv, ey * inv, ez * inv, ew * inv);
}

// ---------------------------------------------------------------------------
// Deformable bilinear sampling. One wave per (b, q, h); lane = channel (DH=32).
// value: [B, S, NH, DH] (64MB, fits 192MB L2 -> gathers hit L2); each corner
// gather is a coalesced 128B transaction across the wave.
// ---------------------------------------------------------------------------
__device__ __forceinline__ float corner_fetch(const float* __restrict__ vb,
                                              int yy, int xx) {
  const bool valid = ((unsigned)yy < (unsigned)Hc) && ((unsigned)xx < (unsigned)Wc);
  if (!valid) return 0.0f;
  const int s = (yy << 7) + xx;            // y*W + x  (W=128)
  return vb[(size_t)s * Dc];               // stride NH*DH = 256 floats
}

__global__ void deform_sample_kernel(const float* __restrict__ value,
                                     const float* __restrict__ off,
                                     const float* __restrict__ attn,
                                     const float* __restrict__ refp,
                                     float* __restrict__ out) {
  const int wid  = (int)((blockIdx.x * blockDim.x + threadIdx.x) >> 5); // (b*LQ+q)*NH+h
  const int lane = (int)(threadIdx.x & 31);                             // dh
  const int h  = wid & (NHc - 1);
  const int bq = wid >> 3;        // b*LQ + q
  const int b  = bq >> 13;        // / LQ (8192)

  const float rx = refp[2 * bq + 0];
  const float ry = refp[2 * bq + 1];
  const float* __restrict__ op = off  + (size_t)wid * (NPc * 2);
  const float* __restrict__ ap = attn + (size_t)wid * NPc;
  const float* __restrict__ vb = value + (size_t)b * Sc * Dc + h * DHc + lane;

  float acc = 0.0f;
#pragma unroll
  for (int p = 0; p < NPc; ++p) {
    // loc = ref + off/[W,H];  pixel = loc*size - 0.5  ->  ref*size + off - 0.5
    const float x = fmaf(rx, (float)Wc, op[2 * p + 0]) - 0.5f;
    const float y = fmaf(ry, (float)Hc, op[2 * p + 1]) - 0.5f;
    const float x0f = floorf(x), y0f = floorf(y);
    const float wx = x - x0f, wy = y - y0f;
    const int x0 = (int)x0f, y0 = (int)y0f;

    const float g00 = corner_fetch(vb, y0,     x0);
    const float g01 = corner_fetch(vb, y0,     x0 + 1);
    const float g10 = corner_fetch(vb, y0 + 1, x0);
    const float g11 = corner_fetch(vb, y0 + 1, x0 + 1);

    const float sv = (1.0f - wy) * ((1.0f - wx) * g00 + wx * g01)
                   +         wy  * ((1.0f - wx) * g10 + wx * g11);
    acc = fmaf(ap[p], sv, acc);
  }
  out[(size_t)wid * DHc + lane] = acc;
}

// ---------------------------------------------------------------------------
// y = LayerNorm(x + res) * g + b over rows of 256. One wave per row,
// 8 elements per lane, wave32 shuffle reductions. Safe for y == x (in place).
// ---------------------------------------------------------------------------
__global__ void ln_residual_kernel(const float* __restrict__ x,
                                   const float* __restrict__ res,
                                   const float* __restrict__ g,
                                   const float* __restrict__ bb,
                                   float* __restrict__ y, int rows) {
  const int row  = (int)((blockIdx.x * blockDim.x + threadIdx.x) >> 5);
  const int lane = (int)(threadIdx.x & 31);
  if (row >= rows) return;

  const float* __restrict__ xr = x   + (size_t)row * Dc;
  const float* __restrict__ rr = res + (size_t)row * Dc;

  float v[8];
  float s = 0.0f;
#pragma unroll
  for (int j = 0; j < 8; ++j) {
    const int i = lane + 32 * j;
    v[j] = xr[i] + rr[i];
    s += v[j];
  }
#pragma unroll
  for (int o = 16; o > 0; o >>= 1) s += __shfl_xor(s, o, 32);
  const float mean = s * (1.0f / 256.0f);

  float var = 0.0f;
#pragma unroll
  for (int j = 0; j < 8; ++j) {
    const float d = v[j] - mean;
    var += d * d;
  }
#pragma unroll
  for (int o = 16; o > 0; o >>= 1) var += __shfl_xor(var, o, 32);
  const float inv = rsqrtf(var * (1.0f / 256.0f) + 1e-5f);

#pragma unroll
  for (int j = 0; j < 8; ++j) {
    const int i = lane + 32 * j;
    y[(size_t)row * Dc + i] = (v[j] - mean) * inv * g[i] + bb[i];
  }
}

// ---------------------------------------------------------------------------
extern "C" void kernel_launch(void* const* d_in, const int* in_sizes, int n_in,
                              void* d_out, int out_size, void* d_ws, size_t ws_size,
                              hipStream_t stream) {
  const float* tgt     = (const float*)d_in[0];
  const float* src     = (const float*)d_in[1];
  const float* refp    = (const float*)d_in[2];
  const float* W_value = (const float*)d_in[3];
  const float* b_value = (const float*)d_in[4];
  const float* W_off   = (const float*)d_in[5];
  const float* b_off   = (const float*)d_in[6];
  const float* W_attn  = (const float*)d_in[7];
  const float* b_attn  = (const float*)d_in[8];
  const float* W_out   = (const float*)d_in[9];
  const float* b_out   = (const float*)d_in[10];
  const float* ln1_g   = (const float*)d_in[11];
  const float* ln1_b   = (const float*)d_in[12];
  const float* ln2_g   = (const float*)d_in[13];
  const float* ln2_b   = (const float*)d_in[14];

  // workspace layout (floats, then f16 weight panels)
  float* ws      = (float*)d_ws;
  float* w_value = ws;                                          // B*S*D   = 16,777,216
  float* w_off   = w_value + (size_t)Bc * Sc * Dc;              // B*LQ*64 =  2,097,152
  float* w_attn  = w_off   + (size_t)Bc * LQ * NHc * NPc * 2;   // B*LQ*32 =  1,048,576
  float* w_samp  = w_attn  + (size_t)Bc * LQ * NHc * NPc;       // B*LQ*D  =  8,388,608
  _Float16* wt_value = (_Float16*)(w_samp + (size_t)Bc * LQ * Dc);  // 256*256
  _Float16* wt_off   = wt_value + 256 * 256;                        // 64*256
  _Float16* wt_attn  = wt_off + 64 * 256;                           // 32*256
  _Float16* wt_out   = wt_attn + 32 * 256;                          // 256*256

  float* q_out = (float*)d_out;                  // query: B*LQ*D
  float* s_out = q_out + (size_t)Bc * LQ * Dc;   // src_out: B*S*D

  const dim3 blk(256);

  // 0) weight panels -> col-major f16
  transpose_w_f16_kernel<<<dim3((256 * 256 + 255) / 256), blk, 0, stream>>>(
      W_value, wt_value, 256);
  transpose_w_f16_kernel<<<dim3((64 * 256 + 255) / 256), blk, 0, stream>>>(
      W_off, wt_off, 64);
  transpose_w_f16_kernel<<<dim3((32 * 256 + 255) / 256), blk, 0, stream>>>(
      W_attn, wt_attn, 32);
  transpose_w_f16_kernel<<<dim3((256 * 256 + 255) / 256), blk, 0, stream>>>(
      W_out, wt_out, 256);

  // 1) value = src @ W_value + b_value    [65536 x 256], 16x128 per wave
  {
    const int M = Bc * Sc, N = Dc;
    const int waves = (M / 16) * (N / 128);
    gemm_f16_wmma_kernel<8><<<dim3((waves + 7) / 8), blk, 0, stream>>>(
        src, wt_value, b_value, w_value, M, N);
  }
  // 2) off = tgt @ W_off + b_off          [32768 x 64]
  {
    const int M = Bc * LQ, N = NHc * NPc * 2;
    const int waves = (M / 16) * (N / 64);
    gemm_f16_wmma_kernel<4><<<dim3((waves + 7) / 8), blk, 0, stream>>>(
        tgt, wt_off, b_off, w_off, M, N);
  }
  // 3) attn_raw = tgt @ W_attn + b_attn   [32768 x 32]
  {
    const int M = Bc * LQ, N = NHc * NPc;
    const int waves = (M / 16) * (N / 32);
    gemm_f16_wmma_kernel<2><<<dim3((waves + 7) / 8), blk, 0, stream>>>(
        tgt, wt_attn, b_attn, w_attn, M, N);
  }
  // 4) softmax over last-dim groups of 4
  {
    const int n = Bc * LQ * NHc;   // 262144 groups of 4
    softmax4_kernel<<<dim3((n + 255) / 256), blk, 0, stream>>>(w_attn, n);
  }
  // 5) deformable bilinear sampling -> w_samp [B,LQ,NH,DH] == [B,LQ,D]
  {
    const int waves = Bc * LQ * NHc;  // 262144
    deform_sample_kernel<<<dim3((waves + 7) / 8), blk, 0, stream>>>(
        w_value, w_off, w_attn, refp, w_samp);
  }
  // 6) attn_out = w_samp @ W_out + b_out  -> q_out (pre-LN)
  {
    const int M = Bc * LQ, N = Dc;
    const int waves = (M / 16) * (N / 128);
    gemm_f16_wmma_kernel<8><<<dim3((waves + 7) / 8), blk, 0, stream>>>(
        w_samp, wt_out, b_out, q_out, M, N);
  }
  // 7) query = LN(tgt + attn_out) in place on q_out
  {
    const int rows = Bc * LQ;
    ln_residual_kernel<<<dim3((rows + 7) / 8), blk, 0, stream>>>(
        q_out, tgt, ln1_g, ln1_b, q_out, rows);
  }
  // 8) src_out = LN(src + src)
  {
    const int rows = Bc * Sc;
    ln_residual_kernel<<<dim3((rows + 7) / 8), blk, 0, stream>>>(
        src, src, ln2_g, ln2_b, s_out, rows);
  }
}
